// MultiheadAttention_69509750719113
// MI455X (gfx1250) — compile-verified, matrix operand order checked by reading, not run on hardware
//
#include <hip/hip_runtime.h>
#include <hip/hip_bf16.h>

typedef __attribute__((ext_vector_type(16))) __bf16 v16bf;
typedef __attribute__((ext_vector_type(8)))  __bf16 v8bf;
typedef __attribute__((ext_vector_type(8)))  float  v8f;
typedef __attribute__((ext_vector_type(4)))  int    v4i;

#define HIDDEN 1024
#define NH     16
#define HD     64
#define SEQ    2048
#define BATCH  4
#define NROWS  (BATCH * SEQ)           // 8192
#define ATTN_SCALE 0.125f              // 64^-0.5
#define NEG_BIG (-3.0e38f)

#define AS1 __attribute__((address_space(1)))
#define AS3 __attribute__((address_space(3)))

// gfx1250 async global->LDS DMA (ASYNCcnt-tracked). Guarded: fall back to a
// plain staged copy if this toolchain doesn't declare the builtin.
#if defined(__gfx1250__) && __has_builtin(__builtin_amdgcn_global_load_async_to_lds_b128)
#define HAVE_ASYNC_LDS 1
#else
#define HAVE_ASYNC_LDS 0
#endif

#if __has_builtin(__builtin_amdgcn_s_wait_asynccnt)
#define WAIT_ASYNC(n) __builtin_amdgcn_s_wait_asynccnt(n)
#else
#define WAIT_ASYNC(n) asm volatile("s_wait_asynccnt %0" ::"i"(n) : "memory")
#endif

// ---------------------------------------------------------------------------
// WMMA helpers
// ---------------------------------------------------------------------------
__device__ __forceinline__ v8f wmma_bf16(v16bf a, v16bf b, v8f c) {
    // D = A(16x32 bf16) * B(32x16 bf16) + C(16x16 f32)
    return __builtin_amdgcn_wmma_f32_16x16x32_bf16(
        /*neg_a=*/false, a, /*neg_b=*/false, b,
        /*c_mod=*/(short)0, c, /*reuse_a=*/false, /*reuse_b=*/false);
}

// Per-lane fragment load for a K-contiguous matrix (row-major, ld in elems).
// A fragments: lane holds row (lane&15), K = 16*(lane>>4)..+15.
// B fragments are the mirror on columns when B is stored K-contiguous.
__device__ __forceinline__ v16bf load_frag(const __bf16* base, int ld) {
    const int lane = threadIdx.x & 31;
    const __bf16* p = base + (size_t)(lane & 15) * ld + ((lane >> 4) << 4);
    return *(const v16bf*)p;
}

// ---------------------------------------------------------------------------
// fp32 -> bf16 conversion kernels
// ---------------------------------------------------------------------------
__global__ void cvt_f32_bf16(const float* __restrict__ src,
                             __bf16* __restrict__ dst, int n) {
    int i = blockIdx.x * 256 + threadIdx.x;
    if (i < n) dst[i] = (__bf16)src[i];
}

// W[K][N] fp32 -> WT[N][K] bf16 (make weights K-contiguous per output column)
__global__ void transpose_cvt(const float* __restrict__ W,
                              __bf16* __restrict__ WT, int K, int N) {
    int idx = blockIdx.x * 256 + threadIdx.x;
    if (idx >= K * N) return;
    int k = idx / N;
    int n = idx - k * N;
    WT[(size_t)n * K + k] = (__bf16)W[idx];
}

// ---------------------------------------------------------------------------
// GEMM: C[M,N] = A[M,K](bf16, row-major) * BT[N,K](bf16, K-contig) + bias
// 256 threads = 8 waves; WG computes 128x64 C. Each wave owns a 16x64 strip.
// The 64x32 B tile is staged once per WG into LDS (double-buffered) via
// async-to-LDS DMA so the next tile's transfer overlaps current WMMAs.
// OUT_MODE 0: bf16 output, 1: fp32 output.
// ---------------------------------------------------------------------------
template <int OUT_MODE>
__global__ __launch_bounds__(256) void gemm_wmma(
    const __bf16* __restrict__ A, const __bf16* __restrict__ BT,
    const float* __restrict__ bias, __bf16* __restrict__ out_bf,
    float* __restrict__ out_f, int M, int N, int K) {
    const int lane = threadIdx.x & 31;
    const int wave = threadIdx.x >> 5;
    const int m0 = blockIdx.y * 128 + wave * 16;
    const int n0 = blockIdx.x * 64;

    __shared__ __align__(32) __bf16 Bs[2][64][32];  // [buf][n][k], 8 KB

    // One b128 chunk per thread: 64 rows x 32 k = 256 chunks of 8 bf16.
    auto stage_B = [&](int sbuf, int kk) {
        const int n = threadIdx.x >> 2;
        const int kc = (threadIdx.x & 3) << 3;
        const __bf16* g = BT + (size_t)(n0 + n) * K + kk + kc;
        __bf16* l = &Bs[sbuf][n][kc];
#if HAVE_ASYNC_LDS
        __builtin_amdgcn_global_load_async_to_lds_b128(
            (AS1 v4i*)g, (AS3 v4i*)l, /*offset=*/0, /*cpol=*/0);
#else
        *(v8bf*)l = *(const v8bf*)g;
#endif
    };

    v8f acc[4] = {v8f{}, v8f{}, v8f{}, v8f{}};

    stage_B(0, 0);
    int buf = 0;
    for (int k0 = 0; k0 < K; k0 += 32) {
        const bool more = (k0 + 32) < K;
        if (more) stage_B(buf ^ 1, k0 + 32);   // overlap next tile's DMA
#if HAVE_ASYNC_LDS
        if (more) WAIT_ASYNC(1); else WAIT_ASYNC(0);  // current tile landed
#endif
        __syncthreads();                        // visible to all 8 waves

        v16bf a = load_frag(A + (size_t)m0 * K + k0, K);
#pragma unroll
        for (int t = 0; t < 4; ++t) {
            v16bf b = load_frag(&Bs[buf][t * 16][0], 32);
            acc[t] = wmma_bf16(a, b, acc[t]);
        }
        __syncthreads();                        // reads done before overwrite
        buf ^= 1;
    }

    // C layout: VGPR r, lanes 0-15 -> M=r, lanes 16-31 -> M=r+8; N = lane&15
    const int mrow = m0 + ((lane >> 4) << 3);
    const int ncol = n0 + (lane & 15);
#pragma unroll
    for (int t = 0; t < 4; ++t) {
        const int n = ncol + t * 16;
        const float bv = bias[n];
#pragma unroll
        for (int r = 0; r < 8; ++r) {
            float v = acc[t][r] + bv;
            size_t off = (size_t)(mrow + r) * N + n;
            if (OUT_MODE)
                out_f[off] = v;
            else
                out_bf[off] = (__bf16)v;
        }
    }
}

// ---------------------------------------------------------------------------
// Flash attention over qkv (bf16 [NROWS][3*HIDDEN]): per (b,h), online-softmax
// over 32-key blocks. 128 threads = 4 waves; wave w owns q rows
// qbase+w*16 .. +15. Writes bf16 context [NROWS][HIDDEN].
// ---------------------------------------------------------------------------
__global__ __launch_bounds__(128) void attn_flash(
    const __bf16* __restrict__ qkv, const int* __restrict__ mask,
    __bf16* __restrict__ ctx) {
    const int lane = threadIdx.x & 31;
    const int wave = threadIdx.x >> 5;
    const int bh = blockIdx.y;
    const int b = bh >> 4;
    const int h = bh & 15;
    const int qbase = blockIdx.x * 64 + wave * 16;
    const size_t rowbase = (size_t)b * SEQ;

    __shared__ __align__(32) __bf16 VT[HD][32];      // V^T tile: [d][key]
    __shared__ __align__(32) __bf16 P[4][16][32];    // per-wave probs [q][key]

    // Q fragments (16 rows x 64 head dims -> two K=32 fragments), kept in regs
    const __bf16* qptr = qkv + (rowbase + qbase) * (3 * HIDDEN) + h * HD;
    v16bf qf0 = load_frag(qptr, 3 * HIDDEN);
    v16bf qf1 = load_frag(qptr + 32, 3 * HIDDEN);

    const int mhalf = ((lane >> 4) << 3);            // row offset for this half
    float mrun[8], lrun[8];
#pragma unroll
    for (int r = 0; r < 8; ++r) { mrun[r] = NEG_BIG; lrun[r] = 0.0f; }
    v8f o[4] = {v8f{}, v8f{}, v8f{}, v8f{}};

    const int* maskrow = mask + ((size_t)b * SEQ + qbase) * SEQ;

    for (int kb = 0; kb < SEQ; kb += 32) {
        __syncthreads();  // previous iteration's LDS reads are done

        // Cooperative stage of V block [kb..kb+32) x [0..HD) transposed
        {
            const int e0 = threadIdx.x * 16;         // 16 elems/thread, 2048 total
            const int kk = e0 >> 6;                  // key within block (const/thread)
            const int d0 = e0 & 63;
            const __bf16* vsrc =
                qkv + (rowbase + kb + kk) * (3 * HIDDEN) + 2 * HIDDEN + h * HD + d0;
#pragma unroll
            for (int i = 0; i < 16; ++i) VT[d0 + i][kk] = vsrc[i];
        }

        // Scores S = Q * K^T : two 16x16 key tiles, K-dim = 64 (2 steps)
        const __bf16* kptr = qkv + (rowbase + kb) * (3 * HIDDEN) + HIDDEN + h * HD;
        v8f s0 = {}, s1 = {};
        s0 = wmma_bf16(qf0, load_frag(kptr, 3 * HIDDEN), s0);
        s0 = wmma_bf16(qf1, load_frag(kptr + 32, 3 * HIDDEN), s0);
        const __bf16* kptr2 = kptr + (size_t)16 * (3 * HIDDEN);
        s1 = wmma_bf16(qf0, load_frag(kptr2, 3 * HIDDEN), s1);
        s1 = wmma_bf16(qf1, load_frag(kptr2 + 32, 3 * HIDDEN), s1);

        // Masked online softmax. Row r of this half lives in VGPR r across the
        // 16 lanes of the half, so xor-shuffles 8..1 reduce within the row.
#pragma unroll
        for (int r = 0; r < 8; ++r) {
            const size_t qoff = (size_t)(r + mhalf) * SEQ + kb + (lane & 15);
            const int m0i = maskrow[qoff];
            const int m1i = maskrow[qoff + 16];
            const float sc0 = s0[r] * ATTN_SCALE;
            const float sc1 = s1[r] * ATTN_SCALE;
            float mx = fmaxf(m0i ? NEG_BIG : sc0, m1i ? NEG_BIG : sc1);
#pragma unroll
            for (int off = 8; off; off >>= 1) mx = fmaxf(mx, __shfl_xor(mx, off, 32));
            const float mnew = fmaxf(mrun[r], mx);
            const float alpha = __expf(mrun[r] - mnew);
            const float p0 = m0i ? 0.0f : __expf(sc0 - mnew);
            const float p1 = m1i ? 0.0f : __expf(sc1 - mnew);
            float ps = p0 + p1;
#pragma unroll
            for (int off = 8; off; off >>= 1) ps += __shfl_xor(ps, off, 32);
            lrun[r] = lrun[r] * alpha + ps;
            mrun[r] = mnew;
#pragma unroll
            for (int t = 0; t < 4; ++t) o[t][r] *= alpha;
            P[wave][r + mhalf][lane & 15] = (__bf16)p0;
            P[wave][r + mhalf][16 + (lane & 15)] = (__bf16)p1;
        }

        __syncthreads();  // VT + P visible

        // O += P(16x32) * V(32x64): A frag from LDS P, B frags from LDS VT
        v16bf pf = load_frag(&P[wave][0][0], 32);
#pragma unroll
        for (int t = 0; t < 4; ++t) {
            v16bf vf = load_frag(&VT[t * 16][0], 32);
            o[t] = wmma_bf16(pf, vf, o[t]);
        }
    }

    // Finalize: divide by row sum (0 for fully-masked rows) and emit bf16 ctx
#pragma unroll
    for (int r = 0; r < 8; ++r) {
        const float inv = lrun[r] > 0.0f ? 1.0f / lrun[r] : 0.0f;
        const size_t row = rowbase + qbase + r + mhalf;
#pragma unroll
        for (int t = 0; t < 4; ++t) {
            const int d = t * 16 + (lane & 15);
            ctx[row * HIDDEN + h * HD + d] = (__bf16)(o[t][r] * inv);
        }
    }
}

// ---------------------------------------------------------------------------
// Host launcher
// ---------------------------------------------------------------------------
extern "C" void kernel_launch(void* const* d_in, const int* in_sizes, int n_in,
                              void* d_out, int out_size, void* d_ws, size_t ws_size,
                              hipStream_t stream) {
    const float* inp  = (const float*)d_in[0];
    const int*   mask = (const int*)d_in[1];
    const float* Wqkv = (const float*)d_in[2];
    const float* bqkv = (const float*)d_in[3];
    const float* Wout = (const float*)d_in[4];
    const float* bout = (const float*)d_in[5];
    float* out = (float*)d_out;

    char* ws = (char*)d_ws;
    __bf16* inp_bf = (__bf16*)ws;  ws += (size_t)NROWS * HIDDEN * 2;        // 16 MB
    __bf16* WqkvT  = (__bf16*)ws;  ws += (size_t)3 * HIDDEN * HIDDEN * 2;   //  6 MB
    __bf16* WoutT  = (__bf16*)ws;  ws += (size_t)HIDDEN * HIDDEN * 2;       //  2 MB
    __bf16* qkv    = (__bf16*)ws;  ws += (size_t)NROWS * 3 * HIDDEN * 2;    // 48 MB
    __bf16* ctx    = (__bf16*)ws;                                           // 16 MB

    const int n_inp = NROWS * HIDDEN;
    cvt_f32_bf16<<<(n_inp + 255) / 256, 256, 0, stream>>>(inp, inp_bf, n_inp);
    transpose_cvt<<<(HIDDEN * 3 * HIDDEN + 255) / 256, 256, 0, stream>>>(
        Wqkv, WqkvT, HIDDEN, 3 * HIDDEN);
    transpose_cvt<<<(HIDDEN * HIDDEN + 255) / 256, 256, 0, stream>>>(
        Wout, WoutT, HIDDEN, HIDDEN);

    // qkv = inp @ Wqkv + bqkv   (bf16 out)
    gemm_wmma<0><<<dim3(3 * HIDDEN / 64, NROWS / 128), 256, 0, stream>>>(
        inp_bf, WqkvT, bqkv, qkv, nullptr, NROWS, 3 * HIDDEN, HIDDEN);

    // attention -> ctx
    attn_flash<<<dim3(SEQ / 64, BATCH * NH), 128, 0, stream>>>(qkv, mask, ctx);

    // out = ctx @ Wout + bout   (fp32 out)
    gemm_wmma<1><<<dim3(HIDDEN / 64, NROWS / 128), 256, 0, stream>>>(
        ctx, WoutT, bout, nullptr, out, NROWS, HIDDEN, HIDDEN);
}
